// SceneEncoder_30116310680223
// MI455X (gfx1250) — matrix operand order repaired and checked by reading, not compiled, not run on hardware
//
#include <hip/hip_runtime.h>
#include <hip/hip_bf16.h>
#include <stdint.h>

// ---------------------------------------------------------------------------
// SceneEncoder for MI455X (gfx1250).
//   * topk (K=50 of M=2048) per (b,a) via LDS argmin rounds
//   * all 2-layer MLPs: layer1 (din<=8) in fp32 VALU, layer2 (256x256) via
//     v_wmma_f32_16x16x32_f16 (f16 in, f32 accumulate) -- roofline-balanced
//     against the ~310MB of fp32 outputs at 23.3 TB/s.
//   * W2 converted+swizzled once into a contiguous-K B-fragment layout (k = 16*(lane/16) + s)
//     in workspace; each WG stages it to LDS with coalesced 128b copies.
//   * layer2 runs a depth-2 software pipeline on B fragments so WMMAs overlap
//     the ds_load latency instead of s_wait_dscnt 0 before every WMMA.
// ---------------------------------------------------------------------------

typedef __attribute__((ext_vector_type(16))) _Float16 v16h;
typedef __attribute__((ext_vector_type(8)))  float    v8f;
typedef __attribute__((ext_vector_type(4)))  float    v4f;

static constexpr int NB = 64;     // batch
static constexpr int NA = 32;     // agents
static constexpr int NM = 2048;   // map points
static constexpr int NK = 50;     // top-k
static constexpr int NL = 16;     // lights
static constexpr int NS = 16;     // stops
static constexpr int NW = 32;     // walkers
static constexpr int NE = 256;    // embed dim
static constexpr int SLOTS = NK + NL + NS + NW;  // 114

// ------------------------------ top-k --------------------------------------
__global__ __launch_bounds__(256) void topk_kernel(
    const float* __restrict__ mapf, const int* __restrict__ mapm,
    const float* __restrict__ agf, int* __restrict__ idx_out) {
  __shared__ float d2[NM];
  __shared__ float rv[256];
  __shared__ int   ri[256];
  __shared__ int   chosen[NK + 6];
  const int ba = blockIdx.x;       // b*NA + a
  const int b  = ba >> 5;
  const int t  = threadIdx.x;
  const float ax = agf[(size_t)ba * 7 + 0];
  const float ay = agf[(size_t)ba * 7 + 1];
  const float* mf = mapf + (size_t)b * NM * 7;
  const int*   mm = mapm + (size_t)b * NM;
  for (int m = t; m < NM; m += 256) {
    float dx = mf[m * 7 + 0] - ax;
    float dy = mf[m * 7 + 1] - ay;
    d2[m] = (mm[m] != 0) ? __builtin_inff() : (dx * dx + dy * dy);
  }
  __syncthreads();
  for (int k = 0; k < NK; ++k) {
    float bv = __builtin_inff(); int bi = 0x7fffffff;
    for (int m = t; m < NM; m += 256) {
      float v = d2[m];
      if (v < bv) { bv = v; bi = m; }
      else if (v == bv && m < bi) { bi = m; }
    }
    rv[t] = bv; ri[t] = bi;
    __syncthreads();
    for (int s = 128; s > 0; s >>= 1) {
      if (t < s) {
        float v2 = rv[t + s]; int i2 = ri[t + s];
        if (v2 < rv[t] || (v2 == rv[t] && i2 < ri[t])) { rv[t] = v2; ri[t] = i2; }
      }
      __syncthreads();
    }
    if (t == 0) {
      int bi0 = ri[0];
      if (bi0 >= NM || bi0 < 0) bi0 = 0;
      chosen[k] = bi0;
      d2[bi0] = __builtin_inff();
    }
    __syncthreads();
  }
  if (t < NK) idx_out[(size_t)ba * NK + t] = chosen[t];
}

// ------------------------------ prep ---------------------------------------
__device__ inline void rel_pose4(const float* f, const float* ag, float out[4]) {
  float dx = f[0] - ag[0], dy = f[1] - ag[1];
  float c = ag[2], s = ag[3];
  out[0] =  c * dx + s * dy;
  out[1] = -s * dx + c * dy;
  out[2] =  f[2] * c + f[3] * s;
  out[3] =  f[3] * c - f[2] * s;
}

__global__ __launch_bounds__(256) void prep_m2a_kernel(
    const float* __restrict__ mapf, const int* __restrict__ mapm,
    const float* __restrict__ agf, const int* __restrict__ idx,
    float* __restrict__ Xa, float* __restrict__ Xb,
    float* __restrict__ scene_masks) {
  int r = blockIdx.x * 256 + threadIdx.x;
  if (r >= NB * NA * NK) return;
  int k = r % NK; int ba = r / NK; int b = ba >> 5;
  int m = idx[r];
  const float* mp = mapf + ((size_t)b * NM + m) * 7;
  const float* ag = agf + (size_t)ba * 7;
  float p[4]; rel_pose4(mp, ag, p);
  const float* ego = agf + (size_t)(b * NA) * 7;
  float ex = mp[0] - ego[0], ey = mp[1] - ego[1];
  float egod = sqrtf(ex * ex + ey * ey);
  int mmask = mapm[(size_t)b * NM + m];
  float unobs = (mmask != 0 || egod > 1.0f) ? 1.0f : 0.0f;
  float* xa = Xa + (size_t)r * 8;
  xa[0] = p[0]; xa[1] = p[1]; xa[2] = p[2]; xa[3] = p[3];
  xa[4] = unobs; xa[5] = 0.f; xa[6] = 0.f; xa[7] = 0.f;
  float* xb = Xb + (size_t)r * 8;
  for (int j = 0; j < 7; ++j) xb[j] = mp[j];
  xb[7] = 0.f;
  float pn = sqrtf(p[0] * p[0] + p[1] * p[1]);
  scene_masks[(size_t)ba * SLOTS + k] = (mmask != 0 || pn > 1.0f) ? 1.0f : 0.0f;
}

__global__ __launch_bounds__(256) void prep_rel_kernel(
    const float* __restrict__ feat, const int* __restrict__ fmask,
    const float* __restrict__ agf, int N, int F,
    float* __restrict__ Xa, float* __restrict__ Xb,
    float* __restrict__ scene_masks, int slot) {
  int r = blockIdx.x * 256 + threadIdx.x;
  if (r >= NB * NA * N) return;
  int n = r % N; int ba = r / N; int b = ba >> 5;
  const float* fp = feat + ((size_t)b * N + n) * F;
  const float* ag = agf + (size_t)ba * 7;
  float p[4]; rel_pose4(fp, ag, p);
  float* xa = Xa + (size_t)r * 8;
  xa[0] = p[0]; xa[1] = p[1]; xa[2] = p[2]; xa[3] = p[3];
  xa[4] = 0.f; xa[5] = 0.f; xa[6] = 0.f; xa[7] = 0.f;
  float* xb = Xb + (size_t)r * 8;
  for (int j = 0; j < 8; ++j) xb[j] = (j < F) ? fp[j] : 0.f;
  float pn = sqrtf(p[0] * p[0] + p[1] * p[1]);
  scene_masks[(size_t)ba * SLOTS + slot + n] =
      (fmask[(size_t)b * N + n] != 0 || pn > 1.0f) ? 1.0f : 0.0f;
}

__global__ __launch_bounds__(256) void prep_agents_kernel(
    const float* __restrict__ agf, float* __restrict__ Xego,
    float* __restrict__ Xag) {
  int r = blockIdx.x * 256 + threadIdx.x;
  if (r >= NB * NA) return;
  int b = r >> 5, a = r & 31;
  float v[8];
  for (int j = 0; j < 7; ++j) v[j] = agf[(size_t)r * 7 + j];
  v[7] = 0.f;
  float* dst = (a == 0) ? (Xego + (size_t)b * 8)
                        : (Xag + (size_t)(b * 31 + a - 1) * 8);
  for (int j = 0; j < 8; ++j) dst[j] = v[j];
}

__global__ __launch_bounds__(256) void prep_a2a_kernel(
    const float* __restrict__ agf, const int* __restrict__ agm,
    float* __restrict__ X, float* __restrict__ a2a_masks) {
  int r = blockIdx.x * 256 + threadIdx.x;
  if (r >= NB * NA * NA) return;
  int j = r & 31; int bi = r >> 5; int i = bi & 31; int b = bi >> 5;
  const float* fp = agf + (size_t)(b * NA + j) * 7;  // feat = agent j
  const float* ag = agf + (size_t)(b * NA + i) * 7;  // frame = agent i
  float p[4]; rel_pose4(fp, ag, p);
  float* x = X + (size_t)r * 8;
  x[0] = p[0]; x[1] = p[1]; x[2] = p[2]; x[3] = p[3];
  x[4] = 0.f; x[5] = 0.f; x[6] = 0.f; x[7] = 0.f;
  float pn = sqrtf(p[0] * p[0] + p[1] * p[1]);
  a2a_masks[r] = (agm[(size_t)b * NA + j] != 0 || pn > 1.0f) ? 1.0f : 0.0f;
}

// ----------------- W2 convert + swizzle (once per launch) -------------------
__device__ inline uint32_t pack_f16x2(float lo, float hi) {
  union { _Float16 h[2]; uint32_t u; } uu;
  uu.h[0] = (_Float16)lo; uu.h[1] = (_Float16)hi;
  return uu.u;
}

// dst[32768] dwords in B-fragment layout: chunk (kb,nt) holds, per lane,
// 16 consecutive-K f16 of column n=lane%16, K0 = kb*32 + (lane/16)*16.
__global__ __launch_bounds__(256) void swz_w2_kernel(
    const float* __restrict__ W2, uint32_t* __restrict__ dst) {
  int it = blockIdx.x * 256 + threadIdx.x;
  if (it >= 32768) return;
  int k2 = it >> 8; int col = it & 255;
  int K0 = k2 << 1;
  uint32_t p = pack_f16x2(W2[(size_t)K0 * NE + col],
                          W2[(size_t)(K0 + 1) * NE + col]);
  int kb = K0 >> 5, nt = col >> 4, n = col & 15;
  int rem = K0 & 31;
  int lane = ((rem >> 4) << 4) | n;
  int dw = (rem & 15) >> 1;
  dst[((((kb << 4) + nt) * 32 + lane) << 3) + dw] = p;
}

// --------------------------- WMMA MLP worker --------------------------------
// Y[row] = MLPa(Xa[row]) (+ MLPb(Xb[row]) if Xb != null); out row remapped via
// orow = (row/seg)*outer + segoff + row%seg. nrows must be a multiple of 16.
__global__ __launch_bounds__(256, 1) void mlp2_wmma_kernel(
    const float* __restrict__ Xa, const float* __restrict__ W1a,
    const float* __restrict__ b1a, const uint32_t* __restrict__ W2aSw,
    const float* __restrict__ b2a, int dina,
    const float* __restrict__ Xb, const float* __restrict__ W1b,
    const float* __restrict__ b1b, const uint32_t* __restrict__ W2bSw,
    const float* __restrict__ b2b, int dinb,
    float* __restrict__ out, int nrows, int seg, int outer, int segoff) {
  extern __shared__ char smem[];
  const bool hasB = (Xb != nullptr);
  uint32_t* w2aS = (uint32_t*)smem;           // 32768 dwords (128KB), swizzled f16x2
  uint32_t* w2bS = w2aS + 32768;              // valid only when hasB
  float* fbase = (float*)(w2aS + (hasB ? 65536 : 32768));
  float* w1aS = fbase;                        // [256][8] fp32, zero padded
  float* b1aS = fbase + 2048;                 // 256
  float* b2S  = fbase + 2304;                 // 256 (= b2a + b2b)
  float* w1bS = fbase + 2560;                 // [256][8] (hasB)
  float* b1bS = fbase + 4608;                 // 256 (hasB)

  const int tid = threadIdx.x;

  // Coalesced 128-bit copy of the pre-swizzled f16 W2 into LDS.
  for (int it = tid; it < 8192; it += 256)
    ((uint4*)w2aS)[it] = ((const uint4*)W2aSw)[it];
  for (int it = tid; it < 2048; it += 256) {
    int h = it >> 3, j = it & 7;
    w1aS[it] = (j < dina) ? W1a[(size_t)j * NE + h] : 0.f;
  }
  if (tid < 256) {
    b1aS[tid] = b1a[tid];
    b2S[tid] = b2a[tid] + (hasB ? b2b[tid] : 0.f);
  }
  if (hasB) {
    for (int it = tid; it < 8192; it += 256)
      ((uint4*)w2bS)[it] = ((const uint4*)W2bSw)[it];
    for (int it = tid; it < 2048; it += 256) {
      int h = it >> 3, j = it & 7;
      w1bS[it] = (j < dinb) ? W1b[(size_t)j * NE + h] : 0.f;
    }
    if (tid < 256) b1bS[tid] = b1b[tid];
  }
  __syncthreads();

  const int wave = tid >> 5;
  const int lane = tid & 31;
  const int lr = lane & 15;   // A-matrix row within tile (M = lane%16)
  const int hg = lane >> 4;   // lane half-group
  const int ntiles = nrows >> 4;
  const int npass = hasB ? 2 : 1;
  const int tstride = gridDim.x * 8;

  for (int tile = blockIdx.x * 8 + wave; tile < ntiles; tile += tstride) {
    const int row = (tile << 4) + lr;

    // Prefetch next tile's X rows (clamped in-bounds; WGP-scope prefetch is
    // non-speculative on gfx1250, so never prefetch past the buffer).
    {
      int ntile = tile + tstride;
      if (ntile < ntiles) {
        __builtin_prefetch(Xa + (size_t)((ntile << 4) + lr) * 8, 0, 1);
        if (hasB) __builtin_prefetch(Xb + (size_t)((ntile << 4) + lr) * 8, 0, 1);
      }
    }

    // 32-bit remapped output offsets for the 8 D rows this lane owns.
    int obase[8];
#pragma unroll
    for (int v = 0; v < 8; ++v) {
      int rowm = (tile << 4) + (hg << 3) + v;
      obase[v] = ((rowm / seg) * outer + segoff + (rowm % seg)) * NE + lr;
    }

    // Persistent accumulators: one 16x16 f32 D tile per N-tile (128 VGPRs).
    v8f C[16];
#pragma unroll
    for (int nt = 0; nt < 16; ++nt) {
      const float bb = b2S[(nt << 4) + lr];
      C[nt] = (v8f){bb, bb, bb, bb, bb, bb, bb, bb};
    }

    // Two sequential passes (MLP a, then MLP b) share the accumulators so
    // only one set of A fragments (64 VGPRs) is live at any time.
    for (int pass = 0; pass < npass; ++pass) {
      const float*    Xp  = pass ? Xb   : Xa;
      const float*    w1S = pass ? w1bS : w1aS;
      const float*    b1S = pass ? b1bS : b1aS;
      const uint32_t* w2S = pass ? w2bS : w2aS;

      // Layer 1 (fp32), written directly in A-fragment element order.
      v16h Af[8];
      {
        const v4f* xp = (const v4f*)(Xp + (size_t)row * 8);
        const v4f x0 = xp[0], x1 = xp[1];
#pragma unroll
        for (int kb = 0; kb < 8; ++kb) {
          v16h af;
#pragma unroll
          for (int e = 0; e < 16; ++e) {
            const int h = (kb << 5) + (hg << 4) + e;
            const v4f* wp = (const v4f*)(w1S + (h << 3));
            const v4f wA = wp[0], wB = wp[1];
            float acc = b1S[h]
                + x0[0] * wA[0] + x0[1] * wA[1] + x0[2] * wA[2] + x0[3] * wA[3]
                + x1[0] * wB[0] + x1[1] * wB[1] + x1[2] * wB[2] + x1[3] * wB[3];
            af[e] = (_Float16)fmaxf(acc, 0.f);
          }
          Af[kb] = af;
        }
      }

      // Layer 2: 128 WMMAs with a depth-2 B-fragment software pipeline so
      // each WMMA waits on dscnt<=4 instead of 0 (loads overlap matrix ops).
      auto ldb = [&](int i) -> v16h {
        const int nt = i >> 3, kb = i & 7;
        return *(const v16h*)(w2S + ((((kb << 4) + nt) * 32 + lane) << 3));
      };
      v16h bf0 = ldb(0);
      v16h bf1 = ldb(1);
#pragma unroll
      for (int i = 0; i < 128; ++i) {
        const int nt = i >> 3, kb = i & 7;
        v16h bn = bf0;
        if (i + 2 < 128) bn = ldb(i + 2);
        C[nt] = __builtin_amdgcn_wmma_f32_16x16x32_f16(
            false, Af[kb], false, bf0, (short)0, C[nt], false, false);
        bf0 = bf1;
        bf1 = bn;
      }
    }

    // Store all 16 D tiles.
#pragma unroll
    for (int nt = 0; nt < 16; ++nt) {
#pragma unroll
      for (int v = 0; v < 8; ++v)
        out[(size_t)(obase[v] + (nt << 4))] = C[nt][v];
    }
  }
}

// ------------------------------ host ---------------------------------------
extern "C" void kernel_launch(void* const* d_in, const int* in_sizes, int n_in,
                              void* d_out, int out_size, void* d_ws,
                              size_t ws_size, hipStream_t stream) {
  (void)n_in; (void)out_size; (void)ws_size;
  const float* mapf   = (const float*)d_in[0];
  const int*   mapm   = (const int*)d_in[1];
  const float* lightf = (const float*)d_in[2];
  const int*   lightm = (const int*)d_in[3];
  const float* stopf  = (const float*)d_in[4];
  const int*   stopm  = (const int*)d_in[5];
  const float* walkf  = (const float*)d_in[6];
  const int*   walkm  = (const int*)d_in[7];
  const float* agf    = (const float*)d_in[8];
  const int*   agm    = (const int*)d_in[9];

  // Params: canonical order map,m2a,lights,l2a,stops,s2a,walkers,w2a,ego,agents,a2a.
  struct MLPP { const float *W1, *b1, *W2, *b2; int din; };
  MLPP P[11];
  static const int sortedPos[11] = {6, 5, 4, 3, 8, 7, 10, 9, 2, 1, 0};
  bool outerSorted    = (in_sizes[10] == 1024);   // a2a.W1 first if pytree-sorted
  bool innerW2Second  = (in_sizes[11] == 65536);  // (W1,W2,b1,b2) vs (W1,b1,W2,b2)
  for (int mi = 0; mi < 11; ++mi) {
    int p = outerSorted ? sortedPos[mi] : mi;
    int base = 10 + p * 4;
    int iW1 = base;
    int iW2 = innerW2Second ? base + 1 : base + 2;
    int ib1 = innerW2Second ? base + 2 : base + 1;
    int ib2 = base + 3;
    P[mi].W1 = (const float*)d_in[iW1];
    P[mi].b1 = (const float*)d_in[ib1];
    P[mi].W2 = (const float*)d_in[iW2];
    P[mi].b2 = (const float*)d_in[ib2];
    P[mi].din = in_sizes[iW1] / 256;
  }

  // Output regions (floats; bool outputs written as 0.0/1.0).
  float* outp        = (float*)d_out;
  float* scene_emb   = outp;                           // 64*32*114*256
  float* scene_masks = outp + 59768832;                // 64*32*114
  float* agents_emb  = outp + 60002304;                // 64*32*256
  float* a2a_emb     = outp + 60526592;                // 64*32*32*256
  float* a2a_masks   = outp + 77303808;                // 64*32*32

  // Workspace layout (float units).
  float* ws = (float*)d_ws;
  int*   idxbuf = (int*)ws;                 // 102400 ints
  float* Xa_m2a = ws + 102400;              // [102400][8]
  float* Xb_map = ws + 921600;              // [102400][8]
  float* Xa_l   = ws + 1740800;             // [32768][8]
  float* Xb_l   = ws + 2002944;
  float* Xa_s   = ws + 2265088;
  float* Xb_s   = ws + 2527232;
  float* Xa_w   = ws + 2789376;             // [65536][8]
  float* Xb_w   = ws + 3313664;
  float* Xego   = ws + 3837952;             // [64][8]
  float* Xag    = ws + 3838464;             // [1984][8]
  float* Xa2a   = ws + 3854336;             // [65536][8]
  uint32_t* w2swBase = (uint32_t*)(ws + 4378624);  // 11 x 32768 dwords
  uint32_t* W2sw[11];
  for (int i = 0; i < 11; ++i) W2sw[i] = w2swBase + (size_t)i * 32768;

  for (int i = 0; i < 11; ++i)
    swz_w2_kernel<<<128, 256, 0, stream>>>(P[i].W2, W2sw[i]);

  topk_kernel<<<NB * NA, 256, 0, stream>>>(mapf, mapm, agf, idxbuf);
  prep_m2a_kernel<<<(NB * NA * NK + 255) / 256, 256, 0, stream>>>(
      mapf, mapm, agf, idxbuf, Xa_m2a, Xb_map, scene_masks);
  prep_rel_kernel<<<(NB * NA * NL + 255) / 256, 256, 0, stream>>>(
      lightf, lightm, agf, NL, 8, Xa_l, Xb_l, scene_masks, NK);
  prep_rel_kernel<<<(NB * NA * NS + 255) / 256, 256, 0, stream>>>(
      stopf, stopm, agf, NS, 5, Xa_s, Xb_s, scene_masks, NK + NL);
  prep_rel_kernel<<<(NB * NA * NW + 255) / 256, 256, 0, stream>>>(
      walkf, walkm, agf, NW, 6, Xa_w, Xb_w, scene_masks, NK + NL + NS);
  prep_agents_kernel<<<(NB * NA + 255) / 256, 256, 0, stream>>>(agf, Xego, Xag);
  prep_a2a_kernel<<<(NB * NA * NA + 255) / 256, 256, 0, stream>>>(
      agf, agm, Xa2a, a2a_masks);

  auto mlp = [&](const float* XA, const MLPP& pa, const uint32_t* w2a,
                 const float* XB, const MLPP* pb, const uint32_t* w2b,
                 float* ob, int nrows, int seg, int outer, int segoff) {
    int ntiles = nrows / 16;
    int wgs = (ntiles + 15) / 16;
    size_t sh = XB ? (size_t)(65536 * 4 + 4864 * 4)
                   : (size_t)(32768 * 4 + 2560 * 4);
    mlp2_wmma_kernel<<<wgs, 256, sh, stream>>>(
        XA, pa.W1, pa.b1, w2a, pa.b2, pa.din,
        XB, XB ? pb->W1 : nullptr, XB ? pb->b1 : nullptr,
        XB ? w2b : nullptr, XB ? pb->b2 : nullptr, XB ? pb->din : 0,
        ob, nrows, seg, outer, segoff);
  };

  // scene_emb slots: [m2a 0..49 | lights 50..65 | stops 66..81 | walkers 82..113]
  mlp(Xa_m2a, P[1], W2sw[1], Xb_map, &P[0], W2sw[0],
      scene_emb, NB * NA * NK, NK, SLOTS, 0);
  mlp(Xa_l, P[3], W2sw[3], Xb_l, &P[2], W2sw[2],
      scene_emb, NB * NA * NL, NL, SLOTS, NK);
  mlp(Xa_s, P[5], W2sw[5], Xb_s, &P[4], W2sw[4],
      scene_emb, NB * NA * NS, NS, SLOTS, NK + NL);
  mlp(Xa_w, P[7], W2sw[7], Xb_w, &P[6], W2sw[6],
      scene_emb, NB * NA * NW, NW, SLOTS, NK + NL + NS);
  mlp(Xego, P[8], W2sw[8], nullptr, nullptr, nullptr,
      agents_emb, NB, 1, NA, 0);                       // ego -> a=0
  mlp(Xag, P[9], W2sw[9], nullptr, nullptr, nullptr,
      agents_emb, NB * 31, 31, NA, 1);                 // a=1..31
  mlp(Xa2a, P[10], W2sw[10], nullptr, nullptr, nullptr,
      a2a_emb, NB * NA * NA, 1 << 30, 0, 0);
}
